// HierarchicalVQ_55748675502204
// MI455X (gfx1250) — matrix operand order repaired and checked by reading, not compiled, not run on hardware
//
#include <hip/hip_runtime.h>
#include <hip/hip_bf16.h>
#include <hip/hip_fp16.h>

// ---------------------------------------------------------------------------
// HierarchicalVQ for MI455X (gfx1250): f16 WMMA GEMMs + TDM (tensor_load_to_lds)
// double-buffered LDS staging with TDM row padding (bank-conflict-free),
// software-pipelined DS->WMMA loops, reuse_a hints on shared-A WMMAs.
// ---------------------------------------------------------------------------

typedef __attribute__((ext_vector_type(16))) _Float16 v16h;
typedef __attribute__((ext_vector_type(8)))  _Float16 h8;
typedef __attribute__((ext_vector_type(4)))  _Float16 h4;
typedef __attribute__((ext_vector_type(8)))  float    v8f;
typedef __attribute__((ext_vector_type(4)))  unsigned v4u;
typedef __attribute__((ext_vector_type(8)))  int      v8i;
typedef __attribute__((ext_vector_type(4)))  int      v4i;

static constexpr int NROWS = 32 * 2048;   // 65536 tokens
static constexpr int DIM   = 512;
static constexpr int DCAT  = 3 * DIM;     // 1536

// Padded LDS row strides (halves): +4 DWORDs per row => 16B-aligned rows and
// lane address stride of 4 banks -> 16 distinct banks per half-wave.
static constexpr int ATS = DIM + 8;       // A tile rows (256 DW + 4 DW pad)
static constexpr int CTS = 16 + 8;        // VQ codebook panel rows (8 DW + 4)
static constexpr int BTS = 64 + 8;        // GEMM B panel rows (32 DW + 4)

#if defined(__HIP_DEVICE_COMPILE__) && defined(__gfx1250__) && \
    __has_builtin(__builtin_amdgcn_tensor_load_to_lds)
#define HAS_TDM 1
#else
#define HAS_TDM 0
#endif

// ----------------------------- WMMA helpers --------------------------------

__device__ __forceinline__ v16h pack16(h8 lo, h8 hi) {
  v16h r;
#pragma unroll
  for (int i = 0; i < 8; ++i) { r[i] = lo[i]; r[i + 8] = hi[i]; }
  return r;
}

// A fragment (16x32, M x K) from row-major f16 global memory.
__device__ __forceinline__ v16h load_a(const _Float16* __restrict__ A, int lda,
                                       int row0, int k0, int lane) {
  const int m = lane & 15, h = lane >> 4;
  const _Float16* p = A + (size_t)(row0 + m) * lda + (k0 + h * 8);
  h8 lo = *(const h8*)p;
  h8 hi = *(const h8*)(p + 16);
  return pack16(lo, hi);
}

// A fragment from f32 source, converted in-register.
__device__ __forceinline__ v16h load_a(const float* __restrict__ A, int lda,
                                       int row0, int k0, int lane) {
  const int m = lane & 15, h = lane >> 4;
  const float* p = A + (size_t)(row0 + m) * lda + (k0 + h * 8);
  v16h r;
#pragma unroll
  for (int i = 0; i < 8; ++i) { r[i] = (_Float16)p[i]; r[i + 8] = (_Float16)p[i + 16]; }
  return r;
}

// A fragment from a padded LDS tile [16 rows][ats cols].
__device__ __forceinline__ v16h load_a_lds(const _Float16* at, int ats, int k0,
                                           int lane) {
  const int m = lane & 15, h = lane >> 4;
  const _Float16* p = at + m * ats + k0 + h * 8;
  h8 lo = *(const h8*)p;
  h8 hi = *(const h8*)(p + 16);
  return pack16(lo, hi);
}

// B fragment (32x16, K x N) from a padded LDS tile [rows][ldt cols].
// Lane l supplies column N=n0+l%16, K = k0+8h+i (i<8) and k0+16+8h+i-8 (i>=8).
__device__ __forceinline__ v16h load_b_lds(const _Float16* bt, int ldt, int k0,
                                           int n0, int lane) {
  const int n = lane & 15, h = lane >> 4;
  const _Float16* p = bt + (k0 + h * 8) * ldt + n0 + n;
  v16h r;
#pragma unroll
  for (int i = 0; i < 8; ++i) { r[i] = p[i * ldt]; r[i + 8] = p[(i + 16) * ldt]; }
  return r;
}

template <bool REUSEA = false>
__device__ __forceinline__ v8f wmma_f16(v16h a, v16h b, v8f c) {
  return __builtin_amdgcn_wmma_f32_16x16x32_f16(false, a, false, b, (short)0, c,
                                                REUSEA, false);
}

// ----------------------------- TDM helpers ---------------------------------

#if HAS_TDM
__device__ __forceinline__ unsigned lds_off(const void* p) {
  return (unsigned)(unsigned long long)p;   // LDS aperture: low 32 bits select LDS
}

// Issue a 2D TDM tile load: tile (tileX x tileY) of 2-byte elements from a
// row-major tensor with row stride strideX (elements) into LDS at ldsOff.
// padIv/padAm (codes per D# 8.4; -1 = no padding): pad_amount DWORDs inserted
// every 2^(padIv+1) DWORDs -> padded LDS row stride, bank-conflict-free.
__device__ __forceinline__ void tdm_load_2d(unsigned ldsOff, const void* gptr,
                                            unsigned tileX, unsigned tileY,
                                            unsigned tensorX, unsigned tensorY,
                                            unsigned long long strideX,
                                            int padIv, int padAm) {
  unsigned long long ga = (unsigned long long)gptr;
  v4u g0;
  g0[0] = 1u;                                            // count=1, user desc
  g0[1] = ldsOff;                                        // lds_addr
  g0[2] = (unsigned)ga;                                  // global_addr[31:0]
  g0[3] = (unsigned)((ga >> 32) & 0x01FFFFFFu) | (2u << 30);  // addr hi | type=2
  unsigned w0 = 1u << 16;                                // data_size=1 (2 bytes)
  if (padIv >= 0)
    w0 |= (1u << 20) | ((unsigned)padIv << 22) | ((unsigned)padAm << 25);
  v8i g1;
  g1[0] = (int)w0;
  g1[1] = (int)((tensorX & 0xFFFFu) << 16);              // tensor_dim0[15:0]
  g1[2] = (int)((tensorX >> 16) | ((tensorY & 0xFFFFu) << 16));
  g1[3] = (int)((tensorY >> 16) | (tileX << 16));        // tile_dim0
  g1[4] = (int)(tileY & 0xFFFFu);                        // tile_dim1 (tile_dim2=0)
  g1[5] = (int)(strideX & 0xFFFFFFFFu);                  // tensor_dim0_stride lo
  g1[6] = (int)((strideX >> 32) & 0xFFFFu);              // stride hi (dim1 stride=0)
  g1[7] = 0;
  v4i g2 = {0, 0, 0, 0};
  v4i g3 = {0, 0, 0, 0};
  v8i g4 = {0, 0, 0, 0, 0, 0, 0, 0};
  __builtin_amdgcn_tensor_load_to_lds(g0, g1, g2, g3, g4, 0);
}
#endif

// ----------------------------- prep kernels --------------------------------

__global__ void k_zero(float* p, int n) {
  int i = blockIdx.x * blockDim.x + threadIdx.x;
  if (i < n) p[i] = 0.f;
}

// WT[c*ldT + colOff + r] = (f16) W[r*C + c]   (W is [R][C] row-major)
__global__ void k_transpose_f16(const float* __restrict__ W, _Float16* __restrict__ WT,
                                int R, int C, int ldT, int colOff) {
  size_t n = (size_t)R * C;
  for (size_t i = blockIdx.x * (size_t)blockDim.x + threadIdx.x; i < n;
       i += (size_t)gridDim.x * blockDim.x) {
    int r = (int)(i / C), c = (int)(i % C);
    WT[(size_t)c * ldT + colOff + r] = (_Float16)W[i];
  }
}

__global__ void k_bias_cat(const float* __restrict__ tb, const float* __restrict__ eb,
                           const float* __restrict__ db, float* __restrict__ out) {
  int i = blockIdx.x * blockDim.x + threadIdx.x;
  if (i < 512) out[i] = tb[i];
  else if (i < 1024) out[i] = eb[i - 512];
  else if (i < 1536) out[i] = db[i - 1024];
}

__global__ void k_cbnorm(const float* __restrict__ cb, float* __restrict__ norm, int K) {
  int j = blockIdx.x * blockDim.x + threadIdx.x;
  if (j < K) {
    const float* r = cb + (size_t)j * DIM;
    float s = 0.f;
#pragma unroll 8
    for (int k = 0; k < DIM; ++k) s += r[k] * r[k];
    norm[j] = s;
  }
}

// ----------------------------- GEMM kernel ---------------------------------
// C[M,Nc] = A[M,Ka] @ BT[Ka,Nc] + bias.  Block = 4 waves, tile 64x64; the
// 32x64 B k-panel is staged in LDS by the TDM (padded rows), double-buffered.
template <typename TA, typename TC>
__global__ void gemm_wmma(const TA* __restrict__ A, int lda,
                          const _Float16* __restrict__ BT, int ldb,
                          const float* __restrict__ bias,
                          TC* __restrict__ C, int ldc, int Ka) {
  __shared__ _Float16 btile[2][32 * BTS + 16];
  const int lane = threadIdx.x & 31;
  const int wave = threadIdx.x >> 5;
  const int row0 = blockIdx.x * 64 + wave * 16;
  const int col0 = blockIdx.y * 64;

  v8f zero = {};
  v8f acc[4];
#pragma unroll
  for (int t = 0; t < 4; ++t) acc[t] = zero;

  const int nk = Ka / 32;
#if HAS_TDM
  if (wave == 0)
    tdm_load_2d(lds_off(btile[0]), BT + col0, 64, 32, (unsigned)ldb, (unsigned)Ka,
                (unsigned long long)ldb, /*padIv=*/4, /*padAm=*/3);  // 32 DW + 4 DW
#endif
  for (int kc = 0; kc < nk; ++kc) {
#if HAS_TDM
    if (wave == 0) {
      if (kc + 1 < nk) {
        tdm_load_2d(lds_off(btile[(kc + 1) & 1]),
                    BT + (size_t)(kc + 1) * 32 * ldb + col0, 64, 32,
                    (unsigned)ldb, (unsigned)Ka, (unsigned long long)ldb, 4, 3);
        __builtin_amdgcn_s_wait_tensorcnt(1);   // buffer kc complete
      } else {
        __builtin_amdgcn_s_wait_tensorcnt(0);
      }
    }
#else
    {   // cooperative fallback staging (padded rows)
      const _Float16* src = BT + (size_t)kc * 32 * ldb + col0;
      for (int e = threadIdx.x; e < 32 * 16; e += blockDim.x) {
        int rr = e >> 4, cc = (e & 15) * 4;
        *(h4*)&btile[kc & 1][rr * BTS + cc] = *(const h4*)&src[(size_t)rr * ldb + cc];
      }
    }
#endif
    __syncthreads();
    const int k0 = kc * 32;
    if (kc + 1 < nk)
      __builtin_prefetch(A + (size_t)(row0 + (lane & 15)) * lda + k0 + 32, 0, 0);
    v16h a = load_a(A, lda, row0, k0, lane);
    const _Float16* bt = btile[kc & 1];
    // manually unrolled, software-pipelined: next B fragment issued before each
    // WMMA; A operand shared -> reuse_a hint on all but the last.
    v16h b0 = load_b_lds(bt, BTS, 0, 0, lane);
    v16h b1 = load_b_lds(bt, BTS, 0, 16, lane);
    acc[0] = wmma_f16<true>(a, b0, acc[0]);
    v16h b2 = load_b_lds(bt, BTS, 0, 32, lane);
    acc[1] = wmma_f16<true>(a, b1, acc[1]);
    v16h b3 = load_b_lds(bt, BTS, 0, 48, lane);
    acc[2] = wmma_f16<true>(a, b2, acc[2]);
    acc[3] = wmma_f16<false>(a, b3, acc[3]);
    __syncthreads();
  }

  const int mo = (lane >> 4) * 8;   // C layout: lanes 0-15 -> M=v, 16-31 -> M=v+8
  const int n  = lane & 15;
#pragma unroll
  for (int t = 0; t < 4; ++t) {
    int col = col0 + t * 16 + n;
    float bv = bias[col];
#pragma unroll
    for (int v = 0; v < 8; ++v) {
      size_t row = (size_t)(row0 + mo + v);
      C[row * ldc + col] = (TC)(acc[t][v] + bv);
    }
  }
}

// ----------------------------- VQ kernel -----------------------------------
// Per wave: 16 rows. Each wave TDM-loads its own 16x512 A tile into LDS
// (padded rows, no VGPR spills); wave 0 streams the 512x16 codebook panels
// (double buffered). Distances d = ||cb||^2 - 2 f.cb via WMMA with a
// software-pipelined DS->WMMA loop, then lane-shuffle argmin, gather, loss.
__global__ void vq_wmma(const _Float16* __restrict__ A, int lda, int colOff,
                        const _Float16* __restrict__ cbT, const float* __restrict__ cb32,
                        const float* __restrict__ cbnorm, int Kcb,
                        int* __restrict__ idxOut,
                        _Float16* __restrict__ qOut, int ldq, int qColOff, int qAccum,
                        _Float16* __restrict__ residOut,
                        float* __restrict__ lossSlot) {
  __shared__ _Float16 atile[4][16 * ATS + 16];  // ~65 KB (WGP has 320 KB LDS)
  __shared__ _Float16 ctile[2][DIM * CTS + 16]; // ~49 KB
  __shared__ int s_idx[4][16];
  const int lane = threadIdx.x & 31;
  const int wave = threadIdx.x >> 5;
  const int row0 = (blockIdx.x * 4 + wave) * 16;

  float minv[8];
  int   mini[8];
#pragma unroll
  for (int v = 0; v < 8; ++v) { minv[v] = 3.4e38f; mini[v] = 0; }

  const int ntiles = Kcb / 16;
#if HAS_TDM
  // every wave DMA's its own A tile; wave 0 also streams the codebook panel
  tdm_load_2d(lds_off(atile[wave]), A + (size_t)row0 * lda + colOff,
              DIM, 16, DIM, 16, (unsigned long long)lda,
              /*padIv=*/7, /*padAm=*/3);                 // 256 DW + 4 DW
  if (wave == 0)
    tdm_load_2d(lds_off(ctile[0]), cbT, 16, DIM, (unsigned)Kcb, DIM,
                (unsigned long long)Kcb, /*padIv=*/2, /*padAm=*/3);  // 8 DW + 4
  if (wave != 0) __builtin_amdgcn_s_wait_tensorcnt(0);   // own A tile done
#endif
  for (int ti = 0; ti < ntiles; ++ti) {
#if HAS_TDM
    if (wave == 0) {
      if (ti + 1 < ntiles) {
        tdm_load_2d(lds_off(ctile[(ti + 1) & 1]), cbT + (ti + 1) * 16, 16, DIM,
                    (unsigned)Kcb, DIM, (unsigned long long)Kcb, 2, 3);
        __builtin_amdgcn_s_wait_tensorcnt(1);   // A tile + panel ti complete
      } else {
        __builtin_amdgcn_s_wait_tensorcnt(0);
      }
    }
#else
    if (ti == 0) {   // stage this wave's A tile once (padded rows)
      for (int e = lane; e < 16 * (DIM / 8); e += 32) {
        int rr = e / (DIM / 8), cc = (e % (DIM / 8)) * 8;
        *(h8*)&atile[wave][rr * ATS + cc] =
            *(const h8*)&A[(size_t)(row0 + rr) * lda + colOff + cc];
      }
    }
    {   // cooperative panel staging: 512x16 halves (padded rows)
      const _Float16* src = cbT + ti * 16;
      for (int e = threadIdx.x; e < DIM * 4; e += blockDim.x) {
        int rr = e >> 2, cc = (e & 3) * 4;
        *(h4*)&ctile[ti & 1][rr * CTS + cc] = *(const h4*)&src[(size_t)rr * Kcb + cc];
      }
    }
#endif
    __syncthreads();
    const _Float16* at = atile[wave];
    const _Float16* bt = ctile[ti & 1];
    v8f acc = {};
    v16h a = load_a_lds(at, ATS, 0, lane);
    v16h b = load_b_lds(bt, CTS, 0, 0, lane);
#pragma unroll
    for (int k0 = 0; k0 < DIM; k0 += 32) {   // software-pipelined: fetch k+1 first
      v16h a2 = a, b2 = b;
      if (k0 + 32 < DIM) {
        a2 = load_a_lds(at, ATS, k0 + 32, lane);
        b2 = load_b_lds(bt, CTS, k0 + 32, 0, lane);
      }
      acc = wmma_f16(a, b, acc);
      a = a2; b = b2;
    }
    const int col = ti * 16 + (lane & 15);
    const float nv = cbnorm[col];
#pragma unroll
    for (int v = 0; v < 8; ++v) {
      float d = nv - 2.0f * acc[v];
      if (d < minv[v]) { minv[v] = d; mini[v] = col; }
    }
    __syncthreads();
  }

  // reduce (val,idx) across the 16 lanes of each half-wave
#pragma unroll
  for (int v = 0; v < 8; ++v) {
    float mv = minv[v];
    int   mi = mini[v];
#pragma unroll
    for (int off = 8; off > 0; off >>= 1) {
      float ov = __shfl_xor(mv, off, 32);
      int   oi = __shfl_xor(mi, off, 32);
      if (ov < mv || (ov == mv && oi < mi)) { mv = ov; mi = oi; }
    }
    minv[v] = mv; mini[v] = mi;
  }
  if ((lane & 15) == 0) {
    const int mo = (lane >> 4) * 8;
#pragma unroll
    for (int v = 0; v < 8; ++v) {
      s_idx[wave][mo + v] = mini[v];
      idxOut[row0 + mo + v] = mini[v];
    }
  }
  __syncthreads();

  // gather codebook rows, loss, quantized write (in place), residual.
  // feature values come from the LDS A tile (no global re-read).
  float sumsq = 0.f;
  for (int r = 0; r < 16; ++r) {
    const size_t row = (size_t)(row0 + r);
    const int j = s_idx[wave][r];
    const float*    q  = cb32 + (size_t)j * DIM;
    const _Float16* f  = atile[wave] + r * ATS;
    _Float16*       qc = qOut + row * (size_t)ldq + qColOff;
#pragma unroll
    for (int it = 0; it < 4; ++it) {
      const int c = it * 128 + lane * 4;
      float4 qv = *(const float4*)(q + c);
      h4 fv = *(const h4*)(f + c);
      float f0 = (float)fv[0], f1 = (float)fv[1], f2 = (float)fv[2], f3 = (float)fv[3];
      float d0 = qv.x - f0, d1 = qv.y - f1, d2 = qv.z - f2, d3 = qv.w - f3;
      sumsq += d0 * d0 + d1 * d1 + d2 * d2 + d3 * d3;
      h4 qh;
      qh[0] = (_Float16)qv.x; qh[1] = (_Float16)qv.y;
      qh[2] = (_Float16)qv.z; qh[3] = (_Float16)qv.w;
      if (qAccum) {   // residual stage 1: qcomb += q1
        h4 prev = *(const h4*)(qc + c);
#pragma unroll
        for (int u = 0; u < 4; ++u) qh[u] = prev[u] + qh[u];
      }
      *(h4*)(qc + c) = qh;
      if (residOut) { // detail stage 0: resid = feat - q0
        h4 rh;
        rh[0] = (_Float16)(f0 - qv.x); rh[1] = (_Float16)(f1 - qv.y);
        rh[2] = (_Float16)(f2 - qv.z); rh[3] = (_Float16)(f3 - qv.w);
        *(h4*)(residOut + row * DIM + c) = rh;
      }
    }
  }
#pragma unroll
  for (int off = 16; off > 0; off >>= 1) sumsq += __shfl_xor(sumsq, off, 32);
  if (lane == 0) atomicAdd(lossSlot, sumsq);
}

__global__ void k_loss(const float* __restrict__ acc, float* __restrict__ out) {
  if (threadIdx.x == 0 && blockIdx.x == 0) {
    float s = acc[0] + acc[1] + acc[2] + acc[3];
    out[0] = 1.25f * s / ((float)NROWS * (float)DIM);   // (1+BETA)*mean, summed
  }
}

// ----------------------------- launcher ------------------------------------

extern "C" void kernel_launch(void* const* d_in, const int* in_sizes, int n_in,
                              void* d_out, int out_size, void* d_ws, size_t ws_size,
                              hipStream_t stream) {
  (void)in_sizes; (void)n_in; (void)out_size; (void)ws_size;

  const float* x    = (const float*)d_in[0];
  const float* tW   = (const float*)d_in[1];
  const float* tb   = (const float*)d_in[2];
  const float* eW   = (const float*)d_in[3];
  const float* eb   = (const float*)d_in[4];
  const float* dW   = (const float*)d_in[5];
  const float* db   = (const float*)d_in[6];
  const float* oW   = (const float*)d_in[7];
  const float* ob   = (const float*)d_in[8];
  const float* cbT32 = (const float*)d_in[9];
  const float* cbE32 = (const float*)d_in[10];
  const float* cb032 = (const float*)d_in[11];
  const float* cb132 = (const float*)d_in[12];

  // workspace carve-out (256B aligned)
  char* ws = (char*)d_ws;
  size_t off = 0;
  auto take = [&](size_t bytes) -> char* {
    char* p = ws + off;
    off += (bytes + 255) & ~(size_t)255;
    return p;
  };
  _Float16* feat  = (_Float16*)take((size_t)NROWS * DCAT * 2);  // feats -> quantized in place
  _Float16* resid = (_Float16*)take((size_t)NROWS * DIM * 2);
  _Float16* wT3   = (_Float16*)take((size_t)DIM * DCAT * 2);    // [512][1536]
  _Float16* outWT = (_Float16*)take((size_t)DCAT * DIM * 2);    // [1536][512]
  _Float16* cbTt  = (_Float16*)take((size_t)DIM * 512 * 2);
  _Float16* cbTe  = (_Float16*)take((size_t)DIM * 256 * 2);
  _Float16* cbT0  = (_Float16*)take((size_t)DIM * 512 * 2);
  _Float16* cbT1  = (_Float16*)take((size_t)DIM * 512 * 2);
  float* biascat  = (float*)take(1536 * 4);
  float* normT    = (float*)take(512 * 4);
  float* normE    = (float*)take(256 * 4);
  float* norm0    = (float*)take(512 * 4);
  float* norm1    = (float*)take(512 * 4);
  float* lossAcc  = (float*)take(16 * 4);

  // output carve-out: quant f32 [N,512] | 4x idx int32 [N] | loss f32 [1]
  float* quant   = (float*)d_out;
  int*   idxBase = (int*)(quant + (size_t)NROWS * DIM);
  int*   idxT = idxBase;
  int*   idxE = idxBase + NROWS;
  int*   idx0 = idxBase + 2 * NROWS;
  int*   idx1 = idxBase + 3 * NROWS;
  float* lossOut = (float*)(idxBase + 4 * NROWS);

  // ---- prep ----
  k_zero<<<1, 64, 0, stream>>>(lossAcc, 16);
  k_transpose_f16<<<512, 256, 0, stream>>>(tW, wT3, 512, 512, DCAT, 0);
  k_transpose_f16<<<512, 256, 0, stream>>>(eW, wT3, 512, 512, DCAT, 512);
  k_transpose_f16<<<512, 256, 0, stream>>>(dW, wT3, 512, 512, DCAT, 1024);
  k_transpose_f16<<<1536, 256, 0, stream>>>(oW, outWT, 512, 1536, DIM, 0);
  k_transpose_f16<<<512, 256, 0, stream>>>(cbT32, cbTt, 512, 512, 512, 0);
  k_transpose_f16<<<256, 256, 0, stream>>>(cbE32, cbTe, 256, 512, 256, 0);
  k_transpose_f16<<<512, 256, 0, stream>>>(cb032, cbT0, 512, 512, 512, 0);
  k_transpose_f16<<<512, 256, 0, stream>>>(cb132, cbT1, 512, 512, 512, 0);
  k_bias_cat<<<6, 256, 0, stream>>>(tb, eb, db, biascat);
  k_cbnorm<<<2, 256, 0, stream>>>(cbT32, normT, 512);
  k_cbnorm<<<1, 256, 0, stream>>>(cbE32, normE, 256);
  k_cbnorm<<<2, 256, 0, stream>>>(cb032, norm0, 512);
  k_cbnorm<<<2, 256, 0, stream>>>(cb132, norm1, 512);

  // ---- fused 3-way projection: feat[N,1536] = x @ Wcat^T + bcat (f16 out) ----
  dim3 g1(NROWS / 64, DCAT / 64);
  gemm_wmma<float, _Float16><<<g1, 128, 0, stream>>>(x, DIM, wT3, DCAT, biascat,
                                                     feat, DCAT, DIM);

  // ---- VQ stages (quantized values overwrite feat sections in place) ----
  vq_wmma<<<NROWS / 64, 128, 0, stream>>>(feat, DCAT, 0,    cbTt, cbT32, normT, 512,
                                          idxT, feat, DCAT, 0,    0, nullptr, lossAcc + 0);
  vq_wmma<<<NROWS / 64, 128, 0, stream>>>(feat, DCAT, 512,  cbTe, cbE32, normE, 256,
                                          idxE, feat, DCAT, 512,  0, nullptr, lossAcc + 1);
  vq_wmma<<<NROWS / 64, 128, 0, stream>>>(feat, DCAT, 1024, cbT0, cb032, norm0, 512,
                                          idx0, feat, DCAT, 1024, 0, resid,   lossAcc + 2);
  vq_wmma<<<NROWS / 64, 128, 0, stream>>>(resid, DIM, 0,    cbT1, cb132, norm1, 512,
                                          idx1, feat, DCAT, 1024, 1, nullptr, lossAcc + 3);

  // ---- output projection: quant[N,512] = qcomb @ outW^T + out_b (f32 out) ----
  dim3 g3(NROWS / 64, DIM / 64);
  gemm_wmma<_Float16, float><<<g3, 128, 0, stream>>>(feat, DCAT, outWT, DIM, ob,
                                                     quant, DIM, DCAT);

  k_loss<<<1, 32, 0, stream>>>(lossAcc, lossOut);
}
